// Discriminator_90950227460154
// MI455X (gfx1250) — compile-verified, matrix operand order checked by reading, not run on hardware
//
#include <hip/hip_runtime.h>

#define NN 50000
#define NE 800000
#define FD 128

typedef __attribute__((ext_vector_type(16))) _Float16 v16h;
typedef __attribute__((ext_vector_type(8)))  _Float16 v8h;
typedef __attribute__((ext_vector_type(8)))  float    v8f;
typedef __attribute__((ext_vector_type(4)))  int      v4i;

union HV16 { v16h v; v8h h[2]; };

#if defined(__gfx1250__) && __has_builtin(__builtin_amdgcn_global_load_async_to_lds_b128)
#define USE_ASYNC_LDS 1
typedef __attribute__((address_space(1))) v4i* gv4i_p;   // global int4*
typedef __attribute__((address_space(3))) v4i* lv4i_p;   // LDS int4*
#else
#define USE_ASYNC_LDS 0
#endif

__device__ __forceinline__ void wait_asynccnt0() {
#if __has_builtin(__builtin_amdgcn_s_wait_asynccnt)
  __builtin_amdgcn_s_wait_asynccnt(0);
#else
  asm volatile("s_wait_asynccnt 0x0" ::: "memory");
#endif
}

// ---------------------------------------------------------------- utilities
__global__ void k_zero(float* __restrict__ p, int n) {
  int i = blockIdx.x * blockDim.x + threadIdx.x;
  int stride = gridDim.x * blockDim.x;
  for (; i < n; i += stride) p[i] = 0.0f;
}

// deg[d] += 1 for every real edge (self-loop contribution added in k_dinv)
__global__ void k_degree(const int* __restrict__ dst, float* __restrict__ deg) {
  int e = blockIdx.x * blockDim.x + threadIdx.x;
  if (e < NE) unsafeAtomicAdd(&deg[dst[e]], 1.0f);
}

// in place: dinv = rsqrt(deg + 1)   (deg includes self-loop -> always >= 1)
__global__ void k_dinv(float* __restrict__ dv) {
  int i = blockIdx.x * blockDim.x + threadIdx.x;
  if (i < NN) dv[i] = rsqrtf(dv[i] + 1.0f);
}

// ------------------------------------------------- GEMM1: xw = x @ W1  (WMMA)
// One block = 16 rows of x; 8 waves each own one 16-wide column tile.
// Staging: ASYNCcnt-tracked global_load_async_to_lds_b128 DMA of raw f32
// tiles (bypasses VGPRs), then in-LDS f32->f16 convert (W transposed).
// K=128 -> 4 x v_wmma_f32_16x16x32_f16, f32 accumulation.
__global__ __launch_bounds__(256) void k_gemm1(const float* __restrict__ x,
                                               const float* __restrict__ W,
                                               float* __restrict__ xw) {
  __shared__ __align__(16) _Float16 sW[FD * FD]; // sW[n*FD+k] = W[k][n] (transposed)
  __shared__ __align__(16) _Float16 sA[16 * FD]; // sA[m*FD+k]
  const int tid  = threadIdx.x;
  const int row0 = blockIdx.x * 16;

#if USE_ASYNC_LDS
  __shared__ __align__(16) float fW[FD * FD];    // raw f32 staging (async DMA dest)
  __shared__ __align__(16) float fA[16 * FD];
  // 16 B per lane per op; 256 threads cover 1024 floats per sweep
  for (int i = tid * 4; i < FD * FD; i += 256 * 4)
    __builtin_amdgcn_global_load_async_to_lds_b128((gv4i_p)(W + i), (lv4i_p)(fW + i), 0, 0);
  for (int i = tid * 4; i < 16 * FD; i += 256 * 4)
    __builtin_amdgcn_global_load_async_to_lds_b128((gv4i_p)(x + row0 * FD + i),
                                                   (lv4i_p)(fA + i), 0, 0);
  wait_asynccnt0();
  __syncthreads();
  for (int i = tid; i < FD * FD; i += 256) {
    int k = i >> 7, n = i & (FD - 1);
    sW[n * FD + k] = (_Float16)fW[i];
  }
  for (int i = tid; i < 16 * FD; i += 256)
    sA[i] = (_Float16)fA[i];
#else
  for (int i = tid; i < FD * FD; i += 256) {
    int k = i >> 7, n = i & (FD - 1);           // coalesced row-major read of W
    sW[n * FD + k] = (_Float16)W[i];
  }
  for (int i = tid; i < 16 * FD; i += 256) {
    int m = i >> 7, k = i & (FD - 1);
    sA[i] = (_Float16)x[(row0 + m) * FD + k];
  }
#endif
  __syncthreads();

  const int wave = tid >> 5;
  const int lane = tid & 31;
  const int half = lane >> 4;        // 0: K 0-7/16-23, 1: K 8-15/24-31
  const int l16  = lane & 15;
  const int khalf = half * 8;
  const int colBase = wave * 16;

  v8f c = {};
#pragma unroll
  for (int kb = 0; kb < 4; ++kb) {
    const int k0 = kb * 32;
    HV16 a, b;
    // 16-bit A layout: lane row = l16; elems 0-7 and 8-15 are two contiguous
    // 8-element K runs -> two 16B LDS loads each.
    a.h[0] = *(const v8h*)&sA[l16 * FD + k0 + khalf];
    a.h[1] = *(const v8h*)&sA[l16 * FD + k0 + khalf + 16];
    // B: lane column = colBase + l16; same K packing (W stored K-contiguous).
    b.h[0] = *(const v8h*)&sW[(colBase + l16) * FD + k0 + khalf];
    b.h[1] = *(const v8h*)&sW[(colBase + l16) * FD + k0 + khalf + 16];
    c = __builtin_amdgcn_wmma_f32_16x16x32_f16(
        /*neg_a=*/false, a.v, /*neg_b=*/false, b.v,
        /*c_mod=*/(short)0, c, /*reuse_a=*/false, /*reuse_b=*/false);
  }

  // C/D layout: VGPR r -> M = r + 8*half, N = l16
  const int col   = colBase + l16;
  const int rbase = row0 + half * 8;
#pragma unroll
  for (int r = 0; r < 8; ++r)
    xw[(rbase + r) * FD + col] = c[r];
}

// ---------------------------------- edge pass 1: acc[dst,:] += xw[src,:]*norm
// one wave per edge (incl. NN virtual self-loop edges); 4 f32 atomics per lane
__global__ __launch_bounds__(256) void k_edge_feat(const int* __restrict__ src,
                                                   const int* __restrict__ dst,
                                                   const float* __restrict__ dinv,
                                                   const float* __restrict__ xw,
                                                   float* __restrict__ acc) {
  const int gw   = (blockIdx.x * 256 + threadIdx.x) >> 5;  // global wave id
  const int lane = threadIdx.x & 31;
  if (gw >= NE + NN) return;
  if (gw + 4096 < NE) {                       // global_prefetch_b8 of index stream
    __builtin_prefetch(&src[gw + 4096], 0, 1);
    __builtin_prefetch(&dst[gw + 4096], 0, 1);
  }
  int s, d;
  if (gw < NE) { s = src[gw]; d = dst[gw]; }
  else         { s = gw - NE; d = s; }        // self-loop
  const float nrm = dinv[s] * dinv[d];
  const float* xs = xw  + s * FD;
  float*       ad = acc + d * FD;
#pragma unroll
  for (int i = 0; i < 4; ++i) {
    const int j = lane + 32 * i;
    unsafeAtomicAdd(&ad[j], xs[j] * nrm);     // global_atomic_add_f32, no return
  }
}

// h = sigmoid(acc + b1)  (in place)
__global__ void k_bias_sigmoid(float* __restrict__ a, const float* __restrict__ b) {
  int i = blockIdx.x * blockDim.x + threadIdx.x;
  int stride = gridDim.x * blockDim.x;
  for (; i < NN * FD; i += stride) {
    float v = a[i] + b[i & (FD - 1)];
    a[i] = 1.0f / (1.0f + __expf(-v));
  }
}

// ------------------------------------------- GEMM2: z = h @ W2  (128-dot/node)
__global__ __launch_bounds__(256) void k_gemm2(const float* __restrict__ h,
                                               const float* __restrict__ W2,
                                               float* __restrict__ z) {
  const int gw   = (blockIdx.x * 256 + threadIdx.x) >> 5;  // one wave per node
  const int lane = threadIdx.x & 31;
  if (gw >= NN) return;
  const float* hr = h + gw * FD;
  float s = 0.0f;
#pragma unroll
  for (int i = 0; i < 4; ++i) s = fmaf(hr[lane + 32 * i], W2[lane + 32 * i], s);
#pragma unroll
  for (int off = 16; off; off >>= 1) s += __shfl_xor(s, off, 32);  // wave32
  if (lane == 0) z[gw] = s;
}

// edge pass 2: acc2[dst] += z[src]*norm   (one thread per edge)
__global__ void k_edge_scalar(const int* __restrict__ src, const int* __restrict__ dst,
                              const float* __restrict__ dinv, const float* __restrict__ z,
                              float* __restrict__ acc2) {
  int e = blockIdx.x * blockDim.x + threadIdx.x;
  if (e >= NE + NN) return;
  int s, d;
  if (e < NE) { s = src[e]; d = dst[e]; }
  else        { s = e - NE; d = s; }
  unsafeAtomicAdd(&acc2[d], z[s] * dinv[s] * dinv[d]);
}

// out = sigmoid(acc2 + b2)
__global__ void k_out(const float* __restrict__ acc2, const float* __restrict__ b2,
                      float* __restrict__ out) {
  int i = blockIdx.x * blockDim.x + threadIdx.x;
  if (i < NN) {
    float v = acc2[i] + b2[0];
    out[i] = 1.0f / (1.0f + __expf(-v));
  }
}

// ---------------------------------------------------------------------------
extern "C" void kernel_launch(void* const* d_in, const int* in_sizes, int n_in,
                              void* d_out, int out_size, void* d_ws, size_t ws_size,
                              hipStream_t stream) {
  const float* x  = (const float*)d_in[0];   // [NN, FD]
  const int*   ei = (const int*)  d_in[1];   // [2, NE]
  const float* W1 = (const float*)d_in[2];   // [FD, FD]
  const float* b1 = (const float*)d_in[3];   // [FD]
  const float* W2 = (const float*)d_in[4];   // [FD, 1]
  const float* b2 = (const float*)d_in[5];   // [1]
  const int* src = ei;
  const int* dst = ei + NE;

  float* ws   = (float*)d_ws;                // ~51.8 MB total
  float* dinv = ws;                          // NN      (deg -> dinv in place)
  float* xw   = dinv + NN;                   // NN*FD
  float* acc1 = xw + NN * FD;                // NN*FD   (reused as h)
  float* z    = acc1 + NN * FD;              // NN
  float* acc2 = z + NN;                      // NN

  k_zero<<<512, 256, 0, stream>>>(dinv, NN);
  k_zero<<<4096, 256, 0, stream>>>(acc1, NN * FD);
  k_zero<<<256, 256, 0, stream>>>(acc2, NN);

  k_degree<<<(NE + 255) / 256, 256, 0, stream>>>(dst, dinv);
  k_dinv<<<(NN + 255) / 256, 256, 0, stream>>>(dinv);

  k_gemm1<<<NN / 16, 256, 0, stream>>>(x, W1, xw);                 // 3125 blocks
  k_edge_feat<<<(NE + NN) / 8, 256, 0, stream>>>(src, dst, dinv, xw, acc1);
  k_bias_sigmoid<<<4096, 256, 0, stream>>>(acc1, b1);
  k_gemm2<<<(NN * 32 + 255) / 256, 256, 0, stream>>>(acc1, W2, z);
  k_edge_scalar<<<(NE + NN + 255) / 256, 256, 0, stream>>>(src, dst, dinv, z, acc2);
  k_out<<<(NN + 255) / 256, 256, 0, stream>>>(acc2, b2, (float*)d_out);
}
